// BASE_75814762709180
// MI455X (gfx1250) — compile-verified, matrix operand order checked by reading, not run on hardware
//
#include <hip/hip_runtime.h>
#include <hip/hip_bf16.h>
#include <math.h>

#define NN   50000
#define FIN  256
#define HD   128
#define KDIM 64
#define EF   800000
#define ESP  400000
#define NEG_SLOPE 0.2f

typedef __attribute__((ext_vector_type(16))) __bf16 v16bf;
typedef __attribute__((ext_vector_type(8)))  __bf16 v8bf;
typedef __attribute__((ext_vector_type(8)))  float  v8f;

__device__ __forceinline__ void atomicMaxF(float* addr, float val) {
  // init = -inf: nonneg -> signed int max, neg -> unsigned int min
  if (val >= 0.f) atomicMax(reinterpret_cast<int*>(addr), __float_as_int(val));
  else            atomicMin(reinterpret_cast<unsigned int*>(addr), __float_as_uint(val));
}

__global__ void fill_kernel(float* __restrict__ p, float v, long n) {
  long i = (long)blockIdx.x * blockDim.x + threadIdx.x;
  if (i < n) p[i] = v;
}

// -------- vectorized fp32 -> bf16 (8 elems/thread, packs via v_cvt_pk_bf16_f32)
__global__ void cvt_f32_bf16(const float* __restrict__ in, __bf16* __restrict__ out,
                             long n8) {
  long i = (long)blockIdx.x * blockDim.x + threadIdx.x;
  if (i >= n8) return;
  const float4* p = reinterpret_cast<const float4*>(in) + i * 2;
  float4 a = p[0], b = p[1];
  v8bf o;
  o[0] = (__bf16)a.x; o[1] = (__bf16)a.y; o[2] = (__bf16)a.z; o[3] = (__bf16)a.w;
  o[4] = (__bf16)b.x; o[5] = (__bf16)b.y; o[6] = (__bf16)b.z; o[7] = (__bf16)b.w;
  reinterpret_cast<v8bf*>(out)[i] = o;
}

// -------- pack W[Kd,Nd] fp32 into WMMA B-fragment order (bf16), optional relu
// index: ((ntile*ksteps + ks)*32 + lane)*16 + e ; e -> k = ks*32 + (e>>3)*16 + half*8 + (e&7)
__global__ void pack_w_bf16(const float* __restrict__ W, __bf16* __restrict__ out,
                            int Kd, int Nd, int reluW) {
  const int ksteps = Kd >> 5;
  const int total = (Nd >> 4) * ksteps * 32;
  const int t = blockIdx.x * blockDim.x + threadIdx.x;
  if (t >= total) return;
  const int lane  = t & 31;
  const int ks    = (t >> 5) % ksteps;
  const int ntile = (t >> 5) / ksteps;
  const int half  = lane >> 4;
  const int n = (ntile << 4) + (lane & 15);
  __bf16* o = out + (size_t)t * 16;
#pragma unroll
  for (int e = 0; e < 16; ++e) {
    const int k = (ks << 5) + ((e >> 3) << 4) + (half << 3) + (e & 7);
    float w = W[(size_t)k * Nd + n];
    if (reluW) w = w > 0.f ? w : 0.f;
    o[e] = (__bf16)w;
  }
}

__device__ __forceinline__ v16bf ld_frag(const __bf16* p0, const __bf16* p1) {
  v8bf lo = *reinterpret_cast<const v8bf*>(p0);
  v8bf hi = *reinterpret_cast<const v8bf*>(p1);
  return __builtin_shufflevector(lo, hi, 0, 1, 2, 3, 4, 5, 6, 7,
                                 8, 9, 10, 11, 12, 13, 14, 15);
}

// -------- dual WMMA GEMM: outl = A*Bl + bl ; outr = A*Br + br (A shared)
__global__ __launch_bounds__(256) void gemm_dual_wmma(
    const __bf16* __restrict__ Abf,
    const __bf16* __restrict__ Bl, const __bf16* __restrict__ Br,
    const float* __restrict__ bl, const float* __restrict__ br,
    float* __restrict__ outl, float* __restrict__ outr,
    int M, int Kd, int Nd)
{
  const int lane  = threadIdx.x & 31;
  const int wave  = threadIdx.x >> 5;
  const int mtile = blockIdx.x * (blockDim.x >> 5) + wave;
  const int ntile = blockIdx.y;
  if (mtile * 16 >= M) return;               // wave-uniform, EXEC all-ones for WMMA

  const int half = lane >> 4;
  const int mrow = (mtile << 4) + (lane & 15);
  const int ncol = (ntile << 4) + (lane & 15);
  const int ksteps = Kd >> 5;

  const __bf16* arow = Abf + (size_t)mrow * Kd + (half << 3);
  const __bf16* pbl  = Bl + ((size_t)ntile * ksteps * 32 + lane) * 16;
  const __bf16* pbr  = Br + ((size_t)ntile * ksteps * 32 + lane) * 16;

  v8f cl = {}, cr = {};
  for (int ks = 0; ks < ksteps; ++ks) {
    v16bf a  = ld_frag(arow, arow + 16);
    v16bf b0 = ld_frag(pbl, pbl + 8);
    v16bf b1 = ld_frag(pbr, pbr + 8);
    cl = __builtin_amdgcn_wmma_f32_16x16x32_bf16(false, a, false, b0,
                                                 (short)0, cl, false, false);
    cr = __builtin_amdgcn_wmma_f32_16x16x32_bf16(false, a, false, b1,
                                                 (short)0, cr, false, false);
    arow += 32; pbl += 512; pbr += 512;
  }
  const float bvl = bl[ncol], bvr = br[ncol];
  const int mbase = (mtile << 4) + (half << 3);   // lanes 16-31: rows M = v+8
#pragma unroll
  for (int v = 0; v < 8; ++v) {
    outl[(size_t)(mbase + v) * Nd + ncol] = cl[v] + bvl;
    outr[(size_t)(mbase + v) * Nd + ncol] = cr[v] + bvr;
  }
}

// -------- single WMMA GEMM (bias optional)
__global__ __launch_bounds__(256) void gemm_wmma(
    const __bf16* __restrict__ Abf, const __bf16* __restrict__ B,
    const float* __restrict__ bias, float* __restrict__ out,
    int M, int Kd, int Nd)
{
  const int lane  = threadIdx.x & 31;
  const int wave  = threadIdx.x >> 5;
  const int mtile = blockIdx.x * (blockDim.x >> 5) + wave;
  const int ntile = blockIdx.y;
  if (mtile * 16 >= M) return;

  const int half = lane >> 4;
  const int mrow = (mtile << 4) + (lane & 15);
  const int ncol = (ntile << 4) + (lane & 15);
  const int ksteps = Kd >> 5;

  const __bf16* arow = Abf + (size_t)mrow * Kd + (half << 3);
  const __bf16* pb   = B + ((size_t)ntile * ksteps * 32 + lane) * 16;

  v8f c = {};
  for (int ks = 0; ks < ksteps; ++ks) {
    v16bf a = ld_frag(arow, arow + 16);
    v16bf b = ld_frag(pb, pb + 8);
    c = __builtin_amdgcn_wmma_f32_16x16x32_bf16(false, a, false, b,
                                                (short)0, c, false, false);
    arow += 32; pb += 512;
  }
  const float bv = bias ? bias[ncol] : 0.f;
  const int mbase = (mtile << 4) + (half << 3);
#pragma unroll
  for (int v = 0; v < 8; ++v)
    out[(size_t)(mbase + v) * Nd + ncol] = c[v] + bv;
}

// -------- per-edge attention logit: e = att . leakyrelu(xl[src]+xr[dst]); track max[dst]
__global__ __launch_bounds__(256) void edge_logits(
    const float* __restrict__ xl, const float* __restrict__ xr,
    const float* __restrict__ att, const int* __restrict__ ei,
    int E, int n, int Hd, float* __restrict__ e_out, float* __restrict__ nmax)
{
  const int lane = threadIdx.x & 31;
  const int wid  = blockIdx.x * (blockDim.x >> 5) + (threadIdx.x >> 5);
  const int Et = E + n;
  if (wid >= Et) return;
  int src, dst;
  if (wid < E) { src = ei[wid]; dst = ei[E + wid]; } else { src = dst = wid - E; }
  const float* pl = xl + (size_t)src * Hd;
  const float* pr = xr + (size_t)dst * Hd;
  float s = 0.f;
  for (int f = lane; f < Hd; f += 32) {
    float v = pl[f] + pr[f];
    v = v > 0.f ? v : NEG_SLOPE * v;
    s += att[f] * v;
  }
#pragma unroll
  for (int off = 16; off; off >>= 1) s += __shfl_xor(s, off, 32);
  if (lane == 0) {
    e_out[wid] = s;
    atomicMaxF(nmax + dst, s);
  }
}

// -------- ee = exp(e - max[dst]); denom[dst] += ee
__global__ void edge_exp(float* __restrict__ e_buf, const float* __restrict__ nmax,
                         float* __restrict__ denom, const int* __restrict__ ei,
                         int E, int n)
{
  const int i = blockIdx.x * blockDim.x + threadIdx.x;
  const int Et = E + n;
  if (i >= Et) return;
  const int dst = (i < E) ? ei[E + i] : (i - E);
  const float ee = __expf(e_buf[i] - nmax[dst]);
  e_buf[i] = ee;
  atomicAdd(denom + dst, ee);
}

// -------- out[dst] += (ee/denom[dst]) * xl[src]
__global__ __launch_bounds__(256) void edge_aggregate(
    const float* __restrict__ xl, const float* __restrict__ e_buf,
    const float* __restrict__ denom, const int* __restrict__ ei,
    int E, int n, int Hd, float* __restrict__ out)
{
  const int lane = threadIdx.x & 31;
  const int wid  = blockIdx.x * (blockDim.x >> 5) + (threadIdx.x >> 5);
  const int Et = E + n;
  if (wid >= Et) return;
  int src, dst;
  if (wid < E) { src = ei[wid]; dst = ei[E + wid]; } else { src = dst = wid - E; }
  const float a = e_buf[wid] / (denom[dst] + 1e-16f);
  const float* pl = xl + (size_t)src * Hd;
  float* po = out + (size_t)dst * Hd;
  for (int f = lane; f < Hd; f += 32) atomicAdd(po + f, a * pl[f]);
}

// -------- h = elu(x + bias)
__global__ void elu_bias(float* __restrict__ x, const float* __restrict__ bias,
                         int Hd, long n)
{
  long i = (long)blockIdx.x * blockDim.x + threadIdx.x;
  if (i >= n) return;
  const float v = x[i] + bias[i % Hd];
  x[i] = v > 0.f ? v : (__expf(v) - 1.f);
}

// -------- row softmax over K=64 (one wave per node), input = acc + bias2, in place
__global__ __launch_bounds__(256) void softmax64(float* __restrict__ z,
                                                 const float* __restrict__ bias, int n)
{
  const int lane = threadIdx.x & 31;
  const int node = blockIdx.x * (blockDim.x >> 5) + (threadIdx.x >> 5);
  if (node >= n) return;
  float* row = z + (size_t)node * KDIM;
  float v0 = row[lane]      + bias[lane];
  float v1 = row[lane + 32] + bias[lane + 32];
  float m = fmaxf(v0, v1);
#pragma unroll
  for (int off = 16; off; off >>= 1) m = fmaxf(m, __shfl_xor(m, off, 32));
  v0 = __expf(v0 - m); v1 = __expf(v1 - m);
  float s = v0 + v1;
#pragma unroll
  for (int off = 16; off; off >>= 1) s += __shfl_xor(s, off, 32);
  const float inv = 1.f / s;
  row[lane] = v0 * inv;
  row[lane + 32] = v1 * inv;
}

// -------- blur[row] += Xpure[col]   (wave per spatial edge)
__global__ __launch_bounds__(256) void blur_scatter(
    const float* __restrict__ xp, const int* __restrict__ ei,
    int E, float* __restrict__ blur)
{
  const int lane = threadIdx.x & 31;
  const int wid  = blockIdx.x * (blockDim.x >> 5) + (threadIdx.x >> 5);
  if (wid >= E) return;
  const int row = ei[wid], col = ei[E + wid];
  const float* ps = xp + (size_t)col * FIN;
  float* pd = blur + (size_t)row * FIN;
  for (int f = lane; f < FIN; f += 32) atomicAdd(pd + f, ps[f]);
}

// -------- out2 = Xpure + alpha * blur ; thread 0 writes trailing scalar 0
__global__ void finalize(const float* __restrict__ xp, float* __restrict__ out2,
                         const float* __restrict__ alpha_p, float* __restrict__ scalar,
                         long n)
{
  long i = (long)blockIdx.x * blockDim.x + threadIdx.x;
  if (i == 0) *scalar = 0.f;
  if (i >= n) return;
  out2[i] = xp[i] + (*alpha_p) * out2[i];
}

extern "C" void kernel_launch(void* const* d_in, const int* in_sizes, int n_in,
                              void* d_out, int out_size, void* d_ws, size_t ws_size,
                              hipStream_t stream) {
  const float* X     = (const float*)d_in[0];
  const int*   eif   = (const int*)d_in[1];
  const int*   eisp  = (const int*)d_in[2];
  const float* W1l   = (const float*)d_in[3];
  const float* b1l   = (const float*)d_in[4];
  const float* W1r   = (const float*)d_in[5];
  const float* b1r   = (const float*)d_in[6];
  const float* att1  = (const float*)d_in[7];
  const float* bias1 = (const float*)d_in[8];
  const float* W2l   = (const float*)d_in[9];
  const float* b2l   = (const float*)d_in[10];
  const float* W2r   = (const float*)d_in[11];
  const float* b2r   = (const float*)d_in[12];
  const float* att2  = (const float*)d_in[13];
  const float* bias2 = (const float*)d_in[14];
  const float* Mmat  = (const float*)d_in[15];
  const float* alpha = (const float*)d_in[16];

  float* ws = (float*)d_ws;
  float* xl1   = ws;                        // r0: 6.4M floats
  float* xr1   = ws + 6400000L;             // r1: 6.4M
  float* out1  = ws + 12800000L;            // r2: 6.4M (first holds Xbf, then out1/h)
  float* ebuf  = ws + 19200000L;            // 850K
  float* nmax  = ws + 20050000L;            // 50K
  float* denom = ws + 20100000L;            // 50K
  __bf16* Zbf  = (__bf16*)(ws + 20150000L); // 3.2M bf16
  __bf16* pk1l = (__bf16*)(ws + 21750000L); // 32768 bf16
  __bf16* pk1r = pk1l + 32768;              // 32768
  __bf16* pk2l = pk1r + 32768;              // 8192
  __bf16* pk2r = pk2l + 8192;               // 8192
  __bf16* pkM  = pk2r + 8192;               // 16384
  // aliases
  __bf16* Xbf  = (__bf16*)out1;             // 12.8M bf16 == r2
  __bf16* hbf  = (__bf16*)xr1;              // 6.4M bf16 == half of r1
  float*  xl2  = xl1;                       // 3.2M
  float*  xr2  = xl1 + 3200000L;            // 3.2M
  float*  xpure = xr1;                      // 12.8M spans r1+r2 (both dead by then)

  float* Zout = (float*)d_out;              // [NN, KDIM]
  float* out2 = Zout + (long)NN * KDIM;     // [NN, FIN]
  float* scal = Zout + (long)NN * KDIM + (long)NN * FIN;

  const int Et = EF + NN;
  const dim3 blk(256);
  const int wpb = 8;
  const int edgeBlocks = (Et + wpb - 1) / wpb;

  auto cfill = [&](float* p, float v, long n) {
    fill_kernel<<<dim3((unsigned)((n + 255) / 256)), blk, 0, stream>>>(p, v, n);
  };
  auto cvt = [&](const float* in, __bf16* out, long n) {
    long n8 = n / 8;
    cvt_f32_bf16<<<dim3((unsigned)((n8 + 255) / 256)), blk, 0, stream>>>(in, out, n8);
  };
  auto pack = [&](const float* W, __bf16* out, int Kd, int Nd, int relu) {
    int total = (Nd >> 4) * (Kd >> 5) * 32;
    pack_w_bf16<<<(total + 255) / 256, blk, 0, stream>>>(W, out, Kd, Nd, relu);
  };

  // -------- one-time conversions / weight packing --------
  cvt(X, Xbf, (long)NN * FIN);
  pack(W1l, pk1l, FIN, HD, 0);
  pack(W1r, pk1r, FIN, HD, 0);
  pack(W2l, pk2l, HD, KDIM, 0);
  pack(W2r, pk2r, HD, KDIM, 0);
  pack(Mmat, pkM, KDIM, FIN, 1);            // relu(M) folded into pack

  // ---------------- layer 1 ----------------
  cfill(nmax, -INFINITY, NN);
  cfill(denom, 0.f, NN);

  const dim3 g1((NN / 16 + wpb - 1) / wpb, HD / 16);
  gemm_dual_wmma<<<g1, blk, 0, stream>>>(Xbf, pk1l, pk1r, b1l, b1r,
                                         xl1, xr1, NN, FIN, HD);
  cfill(out1, 0.f, (long)NN * HD);          // after GEMM: overwrites Xbf (dead)

  edge_logits<<<edgeBlocks, blk, 0, stream>>>(xl1, xr1, att1, eif, EF, NN, HD, ebuf, nmax);
  edge_exp<<<(Et + 255) / 256, blk, 0, stream>>>(ebuf, nmax, denom, eif, EF, NN);
  edge_aggregate<<<edgeBlocks, blk, 0, stream>>>(xl1, ebuf, denom, eif, EF, NN, HD, out1);
  elu_bias<<<((long)NN * HD + 255) / 256, blk, 0, stream>>>(out1, bias1, HD, (long)NN * HD);

  // ---------------- layer 2 ----------------
  cvt(out1, hbf, (long)NN * HD);            // xr1 fp32 dead -> hbf
  cfill(nmax, -INFINITY, NN);
  cfill(denom, 0.f, NN);
  cfill(Zout, 0.f, (long)NN * KDIM);

  const dim3 g2((NN / 16 + wpb - 1) / wpb, KDIM / 16);
  gemm_dual_wmma<<<g2, blk, 0, stream>>>(hbf, pk2l, pk2r, b2l, b2r,
                                         xl2, xr2, NN, HD, KDIM);

  edge_logits<<<edgeBlocks, blk, 0, stream>>>(xl2, xr2, att2, eif, EF, NN, KDIM, ebuf, nmax);
  edge_exp<<<(Et + 255) / 256, blk, 0, stream>>>(ebuf, nmax, denom, eif, EF, NN);
  edge_aggregate<<<edgeBlocks, blk, 0, stream>>>(xl2, ebuf, denom, eif, EF, NN, KDIM, Zout);
  softmax64<<<(NN + wpb - 1) / wpb, blk, 0, stream>>>(Zout, bias2, NN);

  // ---------------- unmix + spatial blur ----------------
  cvt(Zout, Zbf, (long)NN * KDIM);
  const dim3 g3((NN / 16 + wpb - 1) / wpb, FIN / 16);
  gemm_wmma<<<g3, blk, 0, stream>>>(Zbf, pkM, nullptr, xpure, NN, KDIM, FIN);

  cfill(out2, 0.f, (long)NN * FIN);
  blur_scatter<<<(ESP + wpb - 1) / wpb, blk, 0, stream>>>(xpure, eisp, ESP, out2);
  finalize<<<((long)NN * FIN + 255) / 256, blk, 0, stream>>>(xpure, out2, alpha, scal,
                                                             (long)NN * FIN);
}
